// TemporalAttnWithLstm_49924699849362
// MI455X (gfx1250) — matrix operand order repaired and checked by reading, not compiled, not run on hardware
//
#include <hip/hip_runtime.h>
#include <stdint.h>

// Problem constants
#define Bc   512   // batch
#define Tc   512   // time steps
#define Hc   128   // hidden
#define Fc   128   // feature
#define Ec   64    // attention dim
#define G4H  512   // 4*H
#define KC   256   // F + H (concat K for fused gate GEMM)

typedef __bf16 v16bf __attribute__((ext_vector_type(16)));
typedef __bf16 v8bf  __attribute__((ext_vector_type(8)));
typedef float  v8f   __attribute__((ext_vector_type(8)));

__device__ __forceinline__ __bf16 f2bf(float f) {
    union { float f; uint32_t u; } v; v.f = f;
    uint32_t r = v.u + 0x7FFFu + ((v.u >> 16) & 1u);  // round-to-nearest-even
    uint16_t h = (uint16_t)(r >> 16);
    __bf16 b; __builtin_memcpy(&b, &h, 2); return b;
}
__device__ __forceinline__ float sigm_(float x) { return 1.0f / (1.0f + __expf(-x)); }
__device__ __forceinline__ float tanh_(float x) {
    x = fminf(15.0f, fmaxf(-15.0f, x));
    float e = __expf(2.0f * x);
    return (e - 1.0f) / (e + 1.0f);
}

// ---------------------------------------------------------------------------
// Kernel 0: pack weights.
//  wcat[r][k]  (512 x 256 bf16): k<128 -> W_ih[r][k], else W_hh[r][k-128]
//  bias[r]     (512 f32)       : b_ih[r] + b_hh[r]
//  wab         (8192 bf16)     : W_a^T pre-packed in WMMA B-fragment-linear
//                                layout: idx = ((tn*4+ks)*32 + lane)*16 + i
// ---------------------------------------------------------------------------
__global__ void k_pack(const float* __restrict__ W_ih, const float* __restrict__ W_hh,
                       const float* __restrict__ b_ih, const float* __restrict__ b_hh,
                       const float* __restrict__ W_a,
                       __bf16* __restrict__ wcat, float* __restrict__ bias,
                       __bf16* __restrict__ wab) {
    int g = blockIdx.x * blockDim.x + threadIdx.x;   // grid covers 131072
    if (g < G4H * KC) {
        int r = g >> 8;           // /256
        int k = g & 255;
        float v = (k < Fc) ? W_ih[r * Fc + k] : W_hh[r * Hc + (k - Fc)];
        wcat[g] = f2bf(v);
    }
    if (g < G4H) bias[g] = b_ih[g] + b_hh[g];
    if (g < 4 * 4 * 32 * 16) {    // 8192 WaT fragment elements
        int i    = g & 15;
        int lane = (g >> 4) & 31;
        int ks   = (g >> 9) & 3;
        int tn   = g >> 11;
        int k = ks * 32 + ((lane < 16) ? 0 : 16) + i;   // 0..127
        int n = tn * 16 + (lane & 15);                  // 0..63
        wab[g] = f2bf(W_a[n * Hc + k]);                 // B[k][n] = W_a[n][k]
    }
}

// ---------------------------------------------------------------------------
// Kernel 1: LSTM recurrence. 32 blocks x 512 threads (16 waves).
// Block owns batch rows [b0, b0+16). Weights live in registers (16 B-frags /
// wave), h/c live in LDS, loop t = 0..511 with no grid sync.
// Software-pipelined x_t prefetch; 4 independent WMMA accumulator chains.
// ---------------------------------------------------------------------------
__global__ void __launch_bounds__(512, 1)
k_lstm(const float* __restrict__ enc, const __bf16* __restrict__ wcat,
       const float* __restrict__ bias, __bf16* __restrict__ hs) {
    __shared__ __align__(16) __bf16 A_lds[16][KC];   // [x | h] per row, 8 KB
    __shared__ float gates[16][G4H];                 // 32 KB
    __shared__ float c_lds[16][Hc];                  // 8 KB

    const int tid  = threadIdx.x;
    const int lane = tid & 31;
    const int wave = tid >> 5;          // 0..15
    const int j    = wave & 7;          // H-column tile (16 cols)
    const int p    = wave >> 3;         // 0: gates {i,f}, 1: gates {g,o}
    const int g0   = (2 * p) * Hc + 16 * j;
    const int g1   = g0 + Hc;
    const int b0   = blockIdx.x * 16;
    const int ln15 = lane & 15;
    const int kb_a = (lane < 16) ? 0 : 16;   // A-fragment K base
    const int kb_b = (lane < 16) ? 0 : 16;   // B-fragment K base

    // Load loop-invariant weight B-fragments into registers (32B each, aligned)
    v16bf bA[8], bB[8];
#pragma unroll
    for (int ks = 0; ks < 8; ++ks) {
        bA[ks] = *(const v16bf*)(wcat + (size_t)(g0 + ln15) * KC + ks * 32 + kb_b);
        bB[ks] = *(const v16bf*)(wcat + (size_t)(g1 + ln15) * KC + ks * 32 + kb_b);
    }

    // init h = 0 (A h-half), c = 0
#pragma unroll
    for (int it = 0; it < 4; ++it) {
        int e = tid + it * 512;
        int m = e >> 7, n = e & 127;
        A_lds[m][Fc + n] = f2bf(0.0f);
        c_lds[m][n] = 0.0f;
    }
    __syncthreads();

    // preload x_0 into registers (software pipeline)
    float xreg[4];
#pragma unroll
    for (int it = 0; it < 4; ++it) {
        int e = tid + it * 512;
        int m = e >> 7, k = e & 127;
        xreg[it] = enc[(size_t)(b0 + m) * Tc * Fc + k];
    }

    for (int t = 0; t < Tc; ++t) {
        // stage x_t (bf16) into A_lds[:, 0:128] from pipelined registers
#pragma unroll
        for (int it = 0; it < 4; ++it) {
            int e = tid + it * 512;
            A_lds[e >> 7][e & 127] = f2bf(xreg[it]);
        }
        __syncthreads();

        // issue next step's x loads now; GEMM + elementwise hide the latency
        {
            int tnx = (t + 1 < Tc) ? (t + 1) : (Tc - 1);
#pragma unroll
            for (int it = 0; it < 4; ++it) {
                int e = tid + it * 512;
                int m = e >> 7, k = e & 127;
                xreg[it] = enc[(size_t)(b0 + m) * Tc * Fc + (size_t)tnx * Fc + k];
            }
        }

        // 4 independent accumulator chains (even/odd K) to avoid WMMA RAW stalls
        v8f a0a, a0b, a1a, a1b;
#pragma unroll
        for (int r = 0; r < 8; ++r) { a0a[r] = 0.f; a0b[r] = 0.f; a1a[r] = 0.f; a1b[r] = 0.f; }

#pragma unroll
        for (int ks = 0; ks < 8; ks += 2) {
            int k0 = ks * 32 + kb_a;
            v8bf lo0 = *(const v8bf*)&A_lds[ln15][k0];
            v8bf hi0 = *(const v8bf*)&A_lds[ln15][k0 + 8];
            v8bf lo1 = *(const v8bf*)&A_lds[ln15][k0 + 32];
            v8bf hi1 = *(const v8bf*)&A_lds[ln15][k0 + 40];
            v16bf ae = __builtin_shufflevector(lo0, hi0, 0,1,2,3,4,5,6,7,8,9,10,11,12,13,14,15);
            v16bf ao = __builtin_shufflevector(lo1, hi1, 0,1,2,3,4,5,6,7,8,9,10,11,12,13,14,15);
            a0a = __builtin_amdgcn_wmma_f32_16x16x32_bf16(false, ae, false, bA[ks],
                                                          (short)0, a0a, false, false);
            a1a = __builtin_amdgcn_wmma_f32_16x16x32_bf16(false, ae, false, bB[ks],
                                                          (short)0, a1a, false, false);
            a0b = __builtin_amdgcn_wmma_f32_16x16x32_bf16(false, ao, false, bA[ks + 1],
                                                          (short)0, a0b, false, false);
            a1b = __builtin_amdgcn_wmma_f32_16x16x32_bf16(false, ao, false, bB[ks + 1],
                                                          (short)0, a1b, false, false);
        }

        // D layout: lanes 0-15 rows 0..7, lanes 16-31 rows 8..15; N = lane&15
        int mrow = (lane >> 4) << 3;
#pragma unroll
        for (int r = 0; r < 8; ++r) {
            gates[mrow + r][g0 + ln15] = a0a[r] + a0b[r];
            gates[mrow + r][g1 + ln15] = a1a[r] + a1b[r];
        }
        __syncthreads();

        // elementwise LSTM cell update; 2 consecutive columns per thread so
        // gate/c LDS traffic merges to b64 and hs writeback is a packed b32
#pragma unroll
        for (int it = 0; it < 2; ++it) {
            int pe = tid + it * 512;       // 0..1023
            int m = pe >> 6;               // 16 rows
            int n = (pe & 63) << 1;        // even column
            float h2[2];
#pragma unroll
            for (int q = 0; q < 2; ++q) {
                int nn = n + q;
                float gi = gates[m][nn]          + bias[nn];
                float gf = gates[m][Hc + nn]     + bias[Hc + nn];
                float gg = gates[m][2 * Hc + nn] + bias[2 * Hc + nn];
                float go = gates[m][3 * Hc + nn] + bias[3 * Hc + nn];
                float ii = sigm_(gi), ff = sigm_(gf), g2 = tanh_(gg), oo = sigm_(go);
                float c = ff * c_lds[m][nn] + ii * g2;
                c_lds[m][nn] = c;
                h2[q] = oo * tanh_(c);
            }
            __bf16 hb0 = f2bf(h2[0]), hb1 = f2bf(h2[1]);
            A_lds[m][Fc + n]     = hb0;    // next-step h (contiguous pair)
            A_lds[m][Fc + n + 1] = hb1;
            union { __bf16 b[2]; uint32_t u; } pk;
            pk.b[0] = hb0; pk.b[1] = hb1;
            *(uint32_t*)&hs[(size_t)(t * Bc + b0 + m) * Hc + n] = pk.u;
        }
        __syncthreads();
    }
}

// ---------------------------------------------------------------------------
// Kernel 2: attention scores. S[row] = u . tanh(W_a h_row + b_a),
// row = t*B + b (linear over all 262144 hidden states).
// One wave per 16 rows; 16 WMMAs; shfl_xor butterfly over the 16-lane halves.
// ---------------------------------------------------------------------------
__global__ void __launch_bounds__(256)
k_score(const __bf16* __restrict__ hs, const __bf16* __restrict__ wab,
        const float* __restrict__ b_a, const float* __restrict__ u,
        float* __restrict__ S) {
    const int lane = threadIdx.x & 31;
    const int wave = threadIdx.x >> 5;                // 0..7
    const int rowbase = (blockIdx.x * 8 + wave) * 16; // 16 rows per wave
    const int ln15 = lane & 15;
    const int kb_a = (lane < 16) ? 0 : 16;

    v8f acc[4];
#pragma unroll
    for (int tn = 0; tn < 4; ++tn)
#pragma unroll
        for (int r = 0; r < 8; ++r) acc[tn][r] = 0.0f;

    const __bf16* hrow = hs + (size_t)(rowbase + ln15) * Hc;
#pragma unroll
    for (int ks = 0; ks < 4; ++ks) {                  // K = 128 in steps of 32
        int k0 = ks * 32 + kb_a;
        v8bf lo = *(const v8bf*)(hrow + k0);
        v8bf hi = *(const v8bf*)(hrow + k0 + 8);
        v16bf a = __builtin_shufflevector(lo, hi, 0,1,2,3,4,5,6,7,8,9,10,11,12,13,14,15);
#pragma unroll
        for (int tn = 0; tn < 4; ++tn) {              // N = 64 as 4 tiles
            v16bf b = *(const v16bf*)(wab + (size_t)((tn * 4 + ks) * 32 + lane) * 16);
            acc[tn] = __builtin_amdgcn_wmma_f32_16x16x32_bf16(false, a, false, b,
                                                              (short)0, acc[tn], false, false);
        }
    }

    // per-lane: tanh(+b_a) * u, summed over this lane's 4 N columns
    float s[8];
#pragma unroll
    for (int r = 0; r < 8; ++r) s[r] = 0.0f;
#pragma unroll
    for (int tn = 0; tn < 4; ++tn) {
        int n = tn * 16 + ln15;
        float bn = b_a[n], un = u[n];
#pragma unroll
        for (int r = 0; r < 8; ++r) s[r] += tanh_(acc[tn][r] + bn) * un;
    }
    // reduce over N within each 16-lane half (wave32 butterfly)
#pragma unroll
    for (int off = 1; off < 16; off <<= 1)
#pragma unroll
        for (int r = 0; r < 8; ++r) s[r] += __shfl_xor(s[r], off, 32);

    if (ln15 == 0) {
        int base = rowbase + ((lane >> 4) << 3);      // lane0: rows 0-7, lane16: 8-15
#pragma unroll
        for (int r = 0; r < 8; ++r) S[base + r] = s[r];
    }
}

// ---------------------------------------------------------------------------
// Kernel 3: softmax over T + weighted sum of enc_out. One block per batch row.
// Bandwidth bound: streams enc_out once (128 MB).
// ---------------------------------------------------------------------------
__global__ void __launch_bounds__(256)
k_attn(const float* __restrict__ enc, const float* __restrict__ S,
       float* __restrict__ out) {
    __shared__ float sc[Tc];
    __shared__ float red[256];
    __shared__ float pacc[2][Fc];
    const int tid = threadIdx.x;
    const int b   = blockIdx.x;

    // reshape-faithful indexing: scores[b][t] = S_linear[b*T + t]
    for (int t = tid; t < Tc; t += 256) sc[t] = S[(size_t)b * Tc + t];
    __syncthreads();

    float lm = fmaxf(sc[tid], sc[tid + 256]);
    red[tid] = lm; __syncthreads();
    for (int s = 128; s > 0; s >>= 1) {
        if (tid < s) red[tid] = fmaxf(red[tid], red[tid + s]);
        __syncthreads();
    }
    float mx = red[0]; __syncthreads();

    float e0 = __expf(sc[tid] - mx), e1 = __expf(sc[tid + 256] - mx);
    sc[tid] = e0; sc[tid + 256] = e1;
    red[tid] = e0 + e1; __syncthreads();
    for (int s = 128; s > 0; s >>= 1) {
        if (tid < s) red[tid] += red[tid + s];
        __syncthreads();
    }
    float inv = 1.0f / red[0]; __syncthreads();

    // out[b][f] = inv * sum_t sc[t] * enc[b][t][f]; 2 threads per f over even/odd t
    int f = tid & 127, part = tid >> 7;
    float acc = 0.0f;
    const float* eb = enc + (size_t)b * Tc * Fc;
    for (int t = part; t < Tc; t += 2) acc += sc[t] * eb[(size_t)t * Fc + f];
    pacc[part][f] = acc;
    __syncthreads();
    if (tid < Fc) out[(size_t)b * Fc + tid] = (pacc[0][tid] + pacc[1][tid]) * inv;
}

// ---------------------------------------------------------------------------
extern "C" void kernel_launch(void* const* d_in, const int* in_sizes, int n_in,
                              void* d_out, int out_size, void* d_ws, size_t ws_size,
                              hipStream_t stream) {
    const float* enc  = (const float*)d_in[0];
    const float* W_ih = (const float*)d_in[1];
    const float* W_hh = (const float*)d_in[2];
    const float* b_ih = (const float*)d_in[3];
    const float* b_hh = (const float*)d_in[4];
    const float* W_a  = (const float*)d_in[5];
    const float* b_a  = (const float*)d_in[6];
    const float* U_a  = (const float*)d_in[7];
    float* out = (float*)d_out;

    char* ws = (char*)d_ws;
    const size_t WCAT_OFF = 0;                                  // 512*256*2 = 256 KB
    const size_t BIAS_OFF = WCAT_OFF + (size_t)G4H * KC * 2;    // 512*4
    const size_t WAB_OFF  = BIAS_OFF + (size_t)G4H * 4;         // 8192*2
    const size_t HS_OFF   = WAB_OFF + (size_t)8192 * 2;         // T*B*H*2 = 64 MB
    const size_t S_OFF    = HS_OFF + (size_t)Tc * Bc * Hc * 2;  // T*B*4 = 1 MB

    __bf16* wcat = (__bf16*)(ws + WCAT_OFF);
    float*  bias = (float*)(ws + BIAS_OFF);
    __bf16* wab  = (__bf16*)(ws + WAB_OFF);
    __bf16* hs   = (__bf16*)(ws + HS_OFF);
    float*  S    = (float*)(ws + S_OFF);

    k_pack<<<dim3(512), dim3(256), 0, stream>>>(W_ih, W_hh, b_ih, b_hh, W_a,
                                                wcat, bias, wab);
    k_lstm<<<dim3(Bc / 16), dim3(512), 0, stream>>>(enc, wcat, bias, hs);
    k_score<<<dim3((Tc * Bc) / 128), dim3(256), 0, stream>>>(hs, wab, b_a, U_a, S);
    k_attn<<<dim3(Bc), dim3(256), 0, stream>>>(enc, S, out);
}
